// LEnsembleFactory_12214886989916
// MI455X (gfx1250) — matrix operand order repaired and checked by reading, not compiled, not run on hardware
//
#include <hip/hip_runtime.h>
#include <math.h>

typedef _Float16 v16h __attribute__((ext_vector_type(16)));
typedef _Float16 v8h  __attribute__((ext_vector_type(8)));
typedef float    v8f  __attribute__((ext_vector_type(8)));

#define DDIM 32

// ---------------------------------------------------------------------------
// Prep: one wave per row.
//  - convert mus row to f16 (WMMA operand), store to workspace
//  - sq[i] = ||f16(mu_i)||^2  (consistent with the f16 Gram -> exact 0 diag)
//  - foc[i] = exp( tanh(mu_i @ W1 + b1) @ W2 + b2 )   (lane j = hidden unit j)
// ---------------------------------------------------------------------------
__global__ void __launch_bounds__(256)
lens_prep_kernel(const float* __restrict__ mus,
                 const float* __restrict__ W1,
                 const float* __restrict__ b1,
                 const float* __restrict__ W2,
                 const float* __restrict__ b2,
                 _Float16* __restrict__ mus_h,
                 float* __restrict__ sq,
                 float* __restrict__ foc,
                 int N) {
  const int lane = threadIdx.x & 31;
  const int wave = threadIdx.x >> 5;
  const int row  = blockIdx.x * 8 + wave;
  if (row >= N) return;
  const float* mrow = mus + (size_t)row * DDIM;

  // f16 conversion + squared norm of the rounded row
  float x = mrow[lane];
  _Float16 h16 = (_Float16)x;
  mus_h[(size_t)row * DDIM + lane] = h16;
  float xr = (float)h16;
  float s = xr * xr;
  #pragma unroll
  for (int m = 16; m >= 1; m >>= 1) s += __shfl_xor(s, m, 32);
  if (lane == 0) sq[row] = s;

  // MLP (fp32, like the reference): lane j owns hidden unit j
  float acc = b1[lane];
  #pragma unroll
  for (int k = 0; k < DDIM; ++k)
    acc = fmaf(mrow[k], W1[k * DDIM + lane], acc);
  float hj = tanhf(acc);
  float v = hj * W2[lane];
  #pragma unroll
  for (int m = 16; m >= 1; m >>= 1) v += __shfl_xor(v, m, 32);
  if (lane == 0) foc[row] = __expf(v + b2[0]);
}

// ---------------------------------------------------------------------------
// Main: each wave computes a 16x64 strip of L.
//   A fragment (16x32 f16) loaded once, 4x { B load, v_wmma_f32_16x16x32_f16,
//   epilogue exp2, store }.  Operands are L2-resident (mus_h = 1 MiB total);
//   the kernel is bounded by the 1 GiB output stream (~46 us @ 23.3 TB/s).
//   Epilogue per element: add, fmac(-2g), max, fma, raw v_exp_f32, store.
//   Diagonal fixup is inside a wave-uniform branch (taken 1/1024 tiles).
// ---------------------------------------------------------------------------
__global__ void __launch_bounds__(256)
lens_main_kernel(const _Float16* __restrict__ mus_h,
                 const float* __restrict__ sq,
                 const float* __restrict__ foc,
                 float* __restrict__ out,
                 float k2, float lpref, int N) {
  const int lane = threadIdx.x & 31;
  const int wave = threadIdx.x >> 5;
  const int n    = lane & 15;   // tile column (B/C/D lane index)
  const int hi   = lane >> 4;   // half-wave selector

  const int i0    = (blockIdx.y * 8 + wave) * 16;  // row-tile base
  const int jbase = blockIdx.x * 64;               // 4 column tiles

  // --- A fragment: 16x32 f16, contiguous-K layout (same order as B, not the ISA table order) ---
  // lane l: row = l&15, K = 16*hi..16*hi+15 (contiguous, matches B)
  const _Float16* arow = mus_h + (size_t)(i0 + n) * DDIM;
  v16h a = *(const v16h*)(arow + 16 * hi);

  // per-half-wave row norms for this i-tile (C row M = r + 8*hi)
  float sqi[8];
  #pragma unroll
  for (int r = 0; r < 8; ++r) sqi[r] = sq[i0 + 8 * hi + r];

  // Diagonal of a 16x16 tile lives at M == n, i.e. global row i0+n for this
  // lane; hoist the (cheap, L2-hit) foc load.
  const float fval  = foc[i0 + n];
  const int   rdiag = n - 8 * hi;      // r index (0..7 or out of range)

  #pragma unroll
  for (int jt = 0; jt < 4; ++jt) {
    const int j0 = jbase + 16 * jt;

    // --- B fragment: 32x16 f16; lane l holds column l&15, K = 16*hi..+15 ---
    const _Float16* bptr = mus_h + (size_t)(j0 + n) * DDIM + 16 * hi;
    v16h b = *(const v16h*)bptr;

    v8f c = {};
    c = __builtin_amdgcn_wmma_f32_16x16x32_f16(
        /*neg_a=*/false, a, /*neg_b=*/false, b,
        /*c_mod=*/(short)0, c, /*reuse_a=*/false, /*reuse_b=*/false);

    const float sqj = sq[j0 + n];

    float vals[8];
    #pragma unroll
    for (int r = 0; r < 8; ++r) {
      float sqd = fmaxf(sqi[r] + sqj - 2.0f * c[r], 0.0f);
      // raw v_exp_f32: argument ~ [-51, -48.7], far from the denormal guard
      // range, so skip the OCML range-fixup sequence.
      vals[r] = __builtin_amdgcn_exp2f(fmaf(k2, sqd, lpref));
    }

    // wave-uniform diagonal fixup (i0 == j0 on 1/1024 of tiles)
    if (i0 == j0) {
      #pragma unroll
      for (int r = 0; r < 8; ++r)
        if (r == rdiag) vals[r] += fval;
    }

    float* prow = out + (size_t)(i0 + 8 * hi) * N + (j0 + n);
    #pragma unroll
    for (int r = 0; r < 8; ++r)
      prow[(size_t)r * N] = vals[r];
  }
}

// ---------------------------------------------------------------------------
extern "C" void kernel_launch(void* const* d_in, const int* in_sizes, int n_in,
                              void* d_out, int out_size, void* d_ws, size_t ws_size,
                              hipStream_t stream) {
  const float* mus = (const float*)d_in[0];
  const float* W1  = (const float*)d_in[1];
  const float* b1  = (const float*)d_in[2];
  const float* W2  = (const float*)d_in[3];
  const float* b2  = (const float*)d_in[4];
  float* out = (float*)d_out;

  const int D = in_sizes[2];            // 32
  const int N = in_sizes[0] / D;        // 16384

  // scalar prefactor: (2*rho)^(d/2) * (2*pi*sigma^2)^((1-2*rho)*d/2)
  const double rho = 0.01, dd = (double)D;
  const double term1 = pow(2.0 * rho, dd * 0.5);
  const double term2 = pow(2.0 * M_PI, (1.0 - 2.0 * rho) * dd * 0.5);
  // fold prefactor and exp into a single exp2: L = 2^(k2*sqd + lpref)
  const float lpref = (float)(log2(term1 * term2));
  const float k2    = (float)(-(rho / 4.0) * 1.4426950408889634073599246810019);

  // workspace: mus_h (N*32 f16 = 1 MiB) | sq (N f32) | foc (N f32)
  char* ws = (char*)d_ws;
  _Float16* mus_h = (_Float16*)ws;
  float* sqv  = (float*)(ws + (size_t)N * DDIM * sizeof(_Float16));
  float* focv = sqv + N;

  lens_prep_kernel<<<N / 8, 256, 0, stream>>>(mus, W1, b1, W2, b2,
                                              mus_h, sqv, focv, N);

  dim3 grid(N / 64, N / 128);           // x: 4 j-tiles per wave, y: 8 waves/block
  lens_main_kernel<<<grid, 256, 0, stream>>>(mus_h, sqv, focv, out,
                                             k2, lpref, N);
}
